// PricePredictionModel_69630009802882
// MI455X (gfx1250) — compile-verified, matrix operand order checked by reading, not run on hardware
//
#include <hip/hip_runtime.h>

// ---------------------------------------------------------------------------
// 2-layer LSTM for MI455X (gfx1250).
//  - bf16 WMMA (v_wmma_f32_16x16x32_bf16) for all GEMMs, f32 cell state.
//  - Persistent kernel per layer: 32 WGs, each owns 16 hidden units (the 4
//    matching i/f/g/o gate columns) and keeps its [64 x K] bf16 weight slice
//    in LDS for all 512 timesteps (CDNA5: 320KB LDS/WGP).
//  - h exchanged through L2 via a double-buffered global bf16 buffer with one
//    atomic grid barrier per timestep (the latency-critical element).
//  - Branch-free native activations; four B fragments held in distinct
//    registers so LDS loads clause up and overlap with WMMA issue.
// Workspace layout (requires ~97MB):
//   [0)        barrier counter
//   [4KB)      h double buffer    2*128*512 bf16     = 256KB
//   [512KB)    x as bf16          128*512*64 bf16    = 8MB
//   [16MB)     layer-0 h sequence 128*512*512 bf16   = 64MB
//   [96MB)     layer-1 final h    128*512 f32        = 256KB
// ---------------------------------------------------------------------------

typedef __attribute__((ext_vector_type(16))) __bf16       v16bf;
typedef __attribute__((ext_vector_type(8)))  float        v8f;
typedef __attribute__((ext_vector_type(4)))  unsigned int u32x4;

#define B_BATCH 128
#define HID     512
#define NWG     32
#define TSTEPS  512

union FragBF { v16bf v; u32x4 q[2]; };

__device__ __forceinline__ unsigned short f2bf(float f) {
  unsigned int u = __float_as_uint(f);
  u += 0x7FFFu + ((u >> 16) & 1u);          // round-to-nearest-even
  return (unsigned short)(u >> 16);
}

// Load one 16x32 bf16 A/B fragment from a row-major [row][k] matrix.
// Caller passes the per-lane element pointer (row*ld + k0 + lane-group K
// delta already applied). Two 16B vector loads per lane per ISA layout.
__device__ __forceinline__ v16bf load_frag(const unsigned short* p) {
  FragBF f;
  f.q[0] = *reinterpret_cast<const u32x4*>(p);
  f.q[1] = *reinterpret_cast<const u32x4*>(p + 16);
  return f.v;
}

// Branch-free activations. v_exp_f32 computes 2^x natively; gfx1250 also has
// native v_tanh_f32.
__device__ __forceinline__ float fast_sigmoid(float x) {
#if __has_builtin(__builtin_amdgcn_exp2f) && __has_builtin(__builtin_amdgcn_rcpf)
  float e = __builtin_amdgcn_exp2f(-1.44269504088896340736f * x);
  return __builtin_amdgcn_rcpf(1.0f + e);
#else
  return 1.0f / (1.0f + __expf(-x));
#endif
}

__device__ __forceinline__ float fast_tanh(float x) {
#if __has_builtin(__builtin_amdgcn_tanhf)
  return __builtin_amdgcn_tanhf(x);
#else
  return 2.0f * fast_sigmoid(2.0f * x) - 1.0f;
#endif
}

// ---------------------------------------------------------------------------
// Persistent LSTM layer kernel (KX = input width, compile-time).
//   gates = bias + x_t @ W_ih^T + h_{t-1} @ W_hh^T      (K = KX + 512)
// WG `wg` owns hidden cols [wg*16, wg*16+16) => gate cols g*512 + wg*16 + n.
// Wave m (0..7) owns batch rows [16m, 16m+16).
// ---------------------------------------------------------------------------
template <int KX>
__global__ void __launch_bounds__(256, 1)
lstm_persistent(const float* __restrict__ W_ih, const float* __restrict__ W_hh,
                const float* __restrict__ b_ih, const float* __restrict__ b_hh,
                const unsigned short* __restrict__ xin, int x_bstride, int x_tstride,
                unsigned short* __restrict__ hbuf, unsigned int* __restrict__ ctr,
                unsigned short* __restrict__ seq_out, float* __restrict__ last_out,
                int T)
{
  constexpr int Ktot = KX + HID;
  constexpr int ldw  = Ktot + 8;             // pad to spread LDS banks
  extern __shared__ unsigned short wlds[];   // [64][ldw] bf16 weight slice

  const int wg   = blockIdx.x;
  const int tid  = threadIdx.x;
  const int lane = tid & 31;
  const int wave = tid >> 5;                 // M-tile index 0..7
  const int colbase = wg * 16;

  // ---- Stage this WG's weight slice into LDS as bf16 (once, reused T steps)
  for (int idx = tid; idx < 64 * Ktot; idx += 256) {
    int r = idx / Ktot;                      // r = gate*16 + n
    int k = idx - r * Ktot;
    int gate = r >> 4;
    int n    = r & 15;
    int gcol = gate * HID + colbase + n;     // row of W_ih / W_hh
    float w = (k < KX) ? W_ih[(size_t)gcol * KX + k]
                       : W_hh[(size_t)gcol * HID + (k - KX)];
    wlds[r * ldw + k] = f2bf(w);
  }
  __syncthreads();

  // ---- Per-lane column, biases (column-only -> pre-seed accumulators)
  const int mycol = colbase + (lane & 15);
  float bias[4];
#pragma unroll
  for (int g = 0; g < 4; ++g)
    bias[g] = b_ih[g * HID + mycol] + b_hh[g * HID + mycol];

  // ---- f32 cell state lives in registers for the whole sequence
  float cst[8];
#pragma unroll
  for (int r = 0; r < 8; ++r) cst[r] = 0.0f;

  const int arow   = wave * 16 + (lane & 15);     // A row (M = lane&15)
  const int kdelta = (lane >> 4) * 8;             // lane-group K offset
  const int crow0  = wave * 16 + (lane >> 4) * 8; // C/D row base (+r)
  const unsigned short* wb = wlds + (lane & 15) * ldw + kdelta;

  for (int t = 0; t < T; ++t) {
    const unsigned short* hprev = hbuf + (size_t)(t & 1) * B_BATCH * HID;
    unsigned short*       hnext = hbuf + (size_t)((t & 1) ^ 1) * B_BATCH * HID;

    v8f acc[4];
#pragma unroll
    for (int g = 0; g < 4; ++g) {
      v8f a;
#pragma unroll
      for (int e = 0; e < 8; ++e) a[e] = bias[g];
      acc[g] = a;
    }

    // -- x segment: K = 0 .. KX
    const unsigned short* xa =
        xin + (size_t)t * x_tstride + (size_t)arow * x_bstride + kdelta;
#pragma unroll 2
    for (int k0 = 0; k0 < KX; k0 += 32) {
      v16bf afrag = load_frag(xa + k0);
      v16bf barr[4];
#pragma unroll
      for (int g = 0; g < 4; ++g)
        barr[g] = load_frag(wb + g * (16 * ldw) + k0);
#pragma unroll
      for (int g = 0; g < 4; ++g)
        acc[g] = __builtin_amdgcn_wmma_f32_16x16x32_bf16(
            false, afrag, false, barr[g], (short)0, acc[g], false, false);
    }

    // -- recurrent segment: K = KX .. KX+512
    const unsigned short* ha = hprev + (size_t)arow * HID + kdelta;
#pragma unroll 2
    for (int k0 = 0; k0 < HID; k0 += 32) {
      v16bf afrag = load_frag(ha + k0);
      v16bf barr[4];
#pragma unroll
      for (int g = 0; g < 4; ++g)
        barr[g] = load_frag(wb + g * (16 * ldw) + KX + k0);
#pragma unroll
      for (int g = 0; g < 4; ++g)
        acc[g] = __builtin_amdgcn_wmma_f32_16x16x32_bf16(
            false, afrag, false, barr[g], (short)0, acc[g], false, false);
    }

    // ---- Gate nonlinearities + state update (C layout: VGPR r, lanes<16 ->
    //      row=16*wave+r, lanes>=16 -> row=16*wave+8+r, col=lane&15)
#pragma unroll
    for (int r = 0; r < 8; ++r) {
      float ig = fast_sigmoid(acc[0][r]);
      float fg = fast_sigmoid(acc[1][r]);
      float gg = fast_tanh(acc[2][r]);
      float og = fast_sigmoid(acc[3][r]);
      float c  = fg * cst[r] + ig * gg;
      cst[r]   = c;
      float h  = og * fast_tanh(c);
      int row  = crow0 + r;
      unsigned short hb = f2bf(h);
      hnext[(size_t)row * HID + mycol] = hb;
      if (seq_out) seq_out[((size_t)row * T + t) * HID + mycol] = hb;
      if (last_out && t == T - 1) last_out[(size_t)row * HID + mycol] = h;
    }

    // ---- Grid-wide barrier: release our h slice, acquire everyone else's
    __threadfence();
    __syncthreads();
    if (tid == 0) {
      atomicAdd(ctr, 1u);
      const unsigned int target = (unsigned int)(NWG * (t + 1));
      while (atomicAdd(ctr, 0u) < target) __builtin_amdgcn_s_sleep(1);
    }
    __syncthreads();
    __threadfence();
  }
}

// ---------------------------------------------------------------------------
__global__ void f32_to_bf16_kernel(const float* __restrict__ in,
                                   unsigned short* __restrict__ out, int n) {
  int i = blockIdx.x * blockDim.x + threadIdx.x;
  if (i < n) out[i] = f2bf(in[i]);
}

__global__ void reset_kernel(unsigned int* ctr, unsigned short* hbuf, int n) {
  int i = blockIdx.x * blockDim.x + threadIdx.x;
  if (i == 0) *ctr = 0u;
  if (i < n) hbuf[i] = 0;
}

// out[b][c] = sum_k h[b][k] * W_out[c][k] + b_out[c]   (128x3, tiny)
__global__ void head_kernel(const float* __restrict__ hlast,
                            const float* __restrict__ W_out,
                            const float* __restrict__ b_out,
                            float* __restrict__ out) {
  __shared__ float red[128][3];
  int b = blockIdx.x;
  int tid = threadIdx.x;  // 128 threads
  float s0 = 0.f, s1 = 0.f, s2 = 0.f;
  for (int k = tid; k < HID; k += 128) {
    float h = hlast[(size_t)b * HID + k];
    s0 += h * W_out[0 * HID + k];
    s1 += h * W_out[1 * HID + k];
    s2 += h * W_out[2 * HID + k];
  }
  red[tid][0] = s0; red[tid][1] = s1; red[tid][2] = s2;
  __syncthreads();
  if (tid < 3) {
    float s = 0.f;
    for (int i = 0; i < 128; ++i) s += red[i][tid];
    out[b * 3 + tid] = s + b_out[tid];
  }
}

// ---------------------------------------------------------------------------
extern "C" void kernel_launch(void* const* d_in, const int* in_sizes, int n_in,
                              void* d_out, int out_size, void* d_ws, size_t ws_size,
                              hipStream_t stream) {
  const float* x     = (const float*)d_in[0];   // [128,512,64]
  const float* W_ih0 = (const float*)d_in[1];   // [2048,64]
  const float* W_hh0 = (const float*)d_in[2];   // [2048,512]
  const float* b_ih0 = (const float*)d_in[3];
  const float* b_hh0 = (const float*)d_in[4];
  const float* W_ih1 = (const float*)d_in[5];   // [2048,512]
  const float* W_hh1 = (const float*)d_in[6];   // [2048,512]
  const float* b_ih1 = (const float*)d_in[7];
  const float* b_hh1 = (const float*)d_in[8];
  const float* W_out = (const float*)d_in[9];   // [3,512]
  const float* b_out = (const float*)d_in[10];

  char* ws = (char*)d_ws;
  unsigned int*   ctr    = (unsigned int*)(ws);
  unsigned short* hbuf   = (unsigned short*)(ws + (4ull << 10));
  unsigned short* x0bf   = (unsigned short*)(ws + (512ull << 10));
  unsigned short* h1seq  = (unsigned short*)(ws + (16ull << 20));
  float*          h2last = (float*)(ws + (96ull << 20));

  // Allow >64KB dynamic LDS (CDNA5 has 320KB/WGP). Idempotent host calls.
  hipFuncSetAttribute((const void*)lstm_persistent<64>,
                      hipFuncAttributeMaxDynamicSharedMemorySize, 160 * 1024);
  hipFuncSetAttribute((const void*)lstm_persistent<512>,
                      hipFuncAttributeMaxDynamicSharedMemorySize, 160 * 1024);

  // x -> bf16
  const int nx = B_BATCH * TSTEPS * 64;
  f32_to_bf16_kernel<<<(nx + 255) / 256, 256, 0, stream>>>(x, x0bf, nx);

  const int nh = 2 * B_BATCH * HID;
  reset_kernel<<<(nh + 255) / 256, 256, 0, stream>>>(ctr, hbuf, nh);

  // Layer 0: K = 64 + 512; LDS slice = 64*(576+8)*2 = 74752 B
  lstm_persistent<64><<<NWG, 256, 64 * (64 + HID + 8) * 2, stream>>>(
      W_ih0, W_hh0, b_ih0, b_hh0,
      x0bf, /*x_bstride=*/TSTEPS * 64, /*x_tstride=*/64,
      hbuf, ctr, /*seq_out=*/h1seq, /*last_out=*/nullptr, TSTEPS);

  reset_kernel<<<(nh + 255) / 256, 256, 0, stream>>>(ctr, hbuf, nh);

  // Layer 1: K = 512 + 512; LDS slice = 64*(1024+8)*2 = 132096 B
  lstm_persistent<512><<<NWG, 256, 64 * (HID + HID + 8) * 2, stream>>>(
      W_ih1, W_hh1, b_ih1, b_hh1,
      h1seq, /*x_bstride=*/TSTEPS * HID, /*x_tstride=*/HID,
      hbuf, ctr, /*seq_out=*/nullptr, /*last_out=*/h2last, TSTEPS);

  head_kernel<<<B_BATCH, 128, 0, stream>>>(h2last, W_out, b_out, (float*)d_out);
}
